// KnowledgeConsistentAttention_7121055777354
// MI455X (gfx1250) — matrix operand order repaired and checked by reading, not compiled, not run on hardware
//
#include <hip/hip_runtime.h>
#include <hip/hip_bf16.h>
#include <stdint.h>

// KnowledgeConsistentAttention forward, fused flash-softmax formulation.
// Pool(3x3) is linear and precedes softmax -> fold into the B operand:
//   S'[p,q] = kern[p,:] . Fpool[:,q],  out = kern^T @ softmax_p(S')
// Streaming softmax over p => the 4x4096x4096 score tensor is never
// materialized. Both GEMMs run on v_wmma_f32_16x16x32_f16. log2(e) is folded
// into Fpool so the softmax runs natively on v_exp_f32 (exp2) with no
// per-element pre-multiply. kern p-blocks are staged into LDS with
// global_load_async_to_lds_b128 (ASYNCcnt) in both [p][c] and [c][p] layouts.

typedef __attribute__((ext_vector_type(16))) _Float16 v16h;
typedef __attribute__((ext_vector_type(8)))  _Float16 v8h;
typedef __attribute__((ext_vector_type(8)))  float    v8f;

#define NB 4
#define NC 64
#define HW 64
#define NP 4096   // = HW*HW, pixels == p == q index space
#define EPSF 1e-7f
#define LOG2E 1.44269504088896340736f

// ---------------- prep kernel 1: normalized per-pixel kernels ---------------
// kernH[b][p][c] and kernT[b][c][p] = half(normalized (F+eps))
__global__ __launch_bounds__(256) void prep_kern(const float* __restrict__ F,
                                                 _Float16* __restrict__ kernH,
                                                 _Float16* __restrict__ kernT) {
  int idx = blockIdx.x * blockDim.x + threadIdx.x;   // b*NP + p
  if (idx >= NB * NP) return;
  int b = idx >> 12, p = idx & (NP - 1);
  const float* f = F + (size_t)b * NC * NP + p;
  float v[NC];
  float s = 0.f;
#pragma unroll
  for (int c = 0; c < NC; ++c) {
    float x = f[c * NP] + EPSF;
    v[c] = x;
    s += x * x;
  }
  float r = rsqrtf(s);
  _Float16* o  = kernH + (size_t)idx * NC;
  _Float16* ot = kernT + (size_t)b * NC * NP + p;
#pragma unroll
  for (int c = 0; c < NC; ++c) {
    _Float16 hv = (_Float16)(v[c] * r);
    o[c] = hv;            // [p][c] contiguous
    ot[c * NP] = hv;      // [c][p] coalesced across threads
  }
}

// ------- prep kernel 2: 3x3 sum pool * log2(e), stored q-major --------------
// Scores computed against this operand live in the log2 domain, so softmax
// needs only raw v_exp_f32 (2^x) and no per-element multiply by log2(e).
__global__ __launch_bounds__(256) void prep_pool(const float* __restrict__ F,
                                                 _Float16* __restrict__ fpoolT) {
  int idx = blockIdx.x * blockDim.x + threadIdx.x;   // (b*NC+c)*NP + q
  if (idx >= NB * NC * NP) return;
  int q = idx & (NP - 1);
  int bc = idx >> 12;
  int b = bc >> 6, c = bc & 63;
  int y = q >> 6, x = q & 63;
  const float* f = F + (size_t)bc * NP;
  float s = 0.f;
#pragma unroll
  for (int dy = -1; dy <= 1; ++dy) {
    int yy = y + dy;
    if (yy < 0 || yy >= HW) continue;
#pragma unroll
    for (int dx = -1; dx <= 1; ++dx) {
      int xx = x + dx;
      if (xx < 0 || xx >= HW) continue;
      s += f[yy * HW + xx];
    }
  }
  fpoolT[((size_t)b * NP + q) * NC + c] = (_Float16)(s * LOG2E);
}

// A-operand gather: 16-bit A 16x32 layout (ISA 7.12.2):
// lane<16: K = khi*8+{0..7} in V0-3, 16+khi*8+{0..7} in V4-7 (khi=lane>>4).
static __device__ inline v16h ldA(const _Float16* rowbase, int lo, int hi) {
  v8h a = *(const v8h*)(rowbase + lo);
  v8h b = *(const v8h*)(rowbase + hi);
  v16h r;
#pragma unroll
  for (int i = 0; i < 8; ++i) { r[i] = a[i]; r[i + 8] = b[i]; }
  return r;
}

static __device__ inline v8f wmma16(v16h a, v16h b, v8f c) {
  return __builtin_amdgcn_wmma_f32_16x16x32_f16(false, a, false, b,
                                                (short)0, c, false, false);
}

// Async DMA of one 16-byte chunk, global -> LDS (tracked by ASYNCcnt).
// LDS generic addresses keep the byte offset in addr[31:0] (ISA 10.2).
static __device__ inline void async_to_lds16(const void* gptr, void* lptr) {
  unsigned lds_off = (unsigned)(uintptr_t)lptr;
  uint64_t gaddr   = (uint64_t)(uintptr_t)gptr;
  asm volatile("global_load_async_to_lds_b128 %0, %1, off"
               :: "v"(lds_off), "v"(gaddr) : "memory");
}

// ---------------- main fused kernel ----------------------------------------
// grid.x = NB*32; 128 threads = 4 wave32; each wave owns a 32-wide q tile.
__global__ __launch_bounds__(128) void attn_main(const _Float16* __restrict__ kernH,
                                                 const _Float16* __restrict__ kernT,
                                                 const _Float16* __restrict__ fpoolT,
                                                 float* __restrict__ out) {
  __shared__ __attribute__((aligned(32))) _Float16 ldsK[32 * 64];    // [p][c]
  __shared__ __attribute__((aligned(32))) _Float16 ldsKT[64 * 32];   // [c][p]
  __shared__ __attribute__((aligned(32))) _Float16 ldsE[4][32 * 32]; // per-wave [q][p]

  const int b      = blockIdx.x >> 5;
  const int qchunk = blockIdx.x & 31;
  const int wave   = threadIdx.x >> 5;
  const int lane   = threadIdx.x & 31;
  const int lan    = lane & 15;
  const int khi    = lane >> 4;       // which interleaved K half this lane holds
  const int q0     = qchunk * 128 + wave * 32;

  const _Float16* kb  = kernH  + (size_t)b * NP * NC;
  const _Float16* ktb = kernT  + (size_t)b * NP * NC;
  const _Float16* fb  = fpoolT + (size_t)b * NP * NC;

  // GEMM1 B operands (Fpool columns), constant over the p loop.
  // B 32x16 layout: lane holds K=khi*8+{0..7} and 16+khi*8+{0..7} of column lan.
  v16h Bf[2][2];
#pragma unroll
  for (int qt = 0; qt < 2; ++qt) {
    const _Float16* fq = fb + (size_t)(q0 + qt * 16 + lan) * NC;
    Bf[qt][0] = ldA(fq, khi * 8, 16 + khi * 8);        // c in [0,32)
    Bf[qt][1] = ldA(fq, 32 + khi * 8, 48 + khi * 8);   // c in [32,64)
  }

  v8f acc[2][4] = {};                 // [q-subtile][c-tile]
  float m[2] = {-1e30f, -1e30f};
  float l[2] = {0.f, 0.f};

  for (int pb = 0; pb < NP / 32; ++pb) {
    // ---- async-DMA the kern p-block into LDS in both layouts ----
#pragma unroll
    for (int r = 0; r < 2; ++r) {
      int t = threadIdx.x + r * 128;               // 256 chunks of 8 halfs
      async_to_lds16((const v8h*)(kb + (size_t)pb * 32 * NC) + t,
                     (v8h*)ldsK + t);
      int c = t >> 2, poff = (t & 3) * 8;          // kernT slice: row c, 8 p's
      async_to_lds16(ktb + (size_t)c * NP + pb * 32 + poff,
                     ldsKT + c * 32 + poff);
      if (pb + 1 < NP / 32) {                      // hint next block into cache
        __builtin_prefetch((const v8h*)(kb + (size_t)(pb + 1) * 32 * NC) + t, 0, 1);
      }
    }
    asm volatile("s_wait_asynccnt 0x0" ::: "memory");
    __syncthreads();

    // ---- GEMM1: 4 S tiles [16p x 16q] x 2 q-subtiles, K=64 = 2x K=32 ----
    v16h a0l = ldA(ldsK + lan * 64,        khi * 8,      16 + khi * 8);
    v16h a0h = ldA(ldsK + lan * 64,        32 + khi * 8, 48 + khi * 8);
    v16h a1l = ldA(ldsK + (16 + lan) * 64, khi * 8,      16 + khi * 8);
    v16h a1h = ldA(ldsK + (16 + lan) * 64, 32 + khi * 8, 48 + khi * 8);
    v8f z = {};
    v8f s[2][2];                        // [q-subtile][p-subtile]
#pragma unroll
    for (int qt = 0; qt < 2; ++qt) {
      s[qt][0] = wmma16(a0l, Bf[qt][0], z);
      s[qt][0] = wmma16(a0h, Bf[qt][1], s[qt][0]);
      s[qt][1] = wmma16(a1l, Bf[qt][0], z);
      s[qt][1] = wmma16(a1h, Bf[qt][1], s[qt][1]);
    }

    // ---- streaming softmax update (log2 domain), per q-subtile ----
    _Float16* ew = &ldsE[wave][0];
#pragma unroll
    for (int qt = 0; qt < 2; ++qt) {
      float bm = fmaxf(s[qt][0][0], s[qt][1][0]);
#pragma unroll
      for (int j = 1; j < 8; ++j) {
        bm = fmaxf(bm, s[qt][0][j]);
        bm = fmaxf(bm, s[qt][1][j]);
      }
      bm = fmaxf(bm, __shfl_xor(bm, 16, 32));   // lane halves = disjoint p rows
      float mn    = fmaxf(m[qt], bm);
      float scale = __builtin_amdgcn_exp2f(m[qt] - mn);
      m[qt] = mn;
      l[qt] *= scale;
#pragma unroll
      for (int ct = 0; ct < 4; ++ct)
#pragma unroll
        for (int j = 0; j < 8; ++j) acc[qt][ct][j] *= scale;

      float ls = 0.f;
      v8h e0, e1;                       // lane's p indices are contiguous!
#pragma unroll
      for (int j = 0; j < 8; ++j) {
        float e = __builtin_amdgcn_exp2f(s[qt][0][j] - mn);
        ls += e; e0[j] = (_Float16)e;
      }
#pragma unroll
      for (int j = 0; j < 8; ++j) {
        float e = __builtin_amdgcn_exp2f(s[qt][1][j] - mn);
        ls += e; e1[j] = (_Float16)e;
      }
      ls += __shfl_xor(ls, 16, 32);
      l[qt] += ls;

      // [q][p] layout: row q, p = 8*khi+{0..7} (tile0), 16+8*khi+{0..7} (tile1)
      _Float16* er = ew + (qt * 16 + lan) * 32;
      *(v8h*)(er + 8 * khi)      = e0;  // one b128 store per tile
      *(v8h*)(er + 16 + 8 * khi) = e1;
    }

    // E as B operand: lane holds K(p)=khi*8+{0..7} and 16+khi*8+{0..7} of column q
    v16h Be0 = ldA(ew + lan * 32, khi * 8, 16 + khi * 8);
    v16h Be1 = ldA(ew + (16 + lan) * 32, khi * 8, 16 + khi * 8);

    // ---- GEMM2: acc += kern^T @ E ; A tiles reused across both q-subtiles
#pragma unroll
    for (int ct = 0; ct < 4; ++ct) {
      v16h a = ldA(ldsKT + (ct * 16 + lan) * 32, khi * 8, 16 + khi * 8);
      acc[0][ct] = wmma16(a, Be0, acc[0][ct]);
      acc[1][ct] = wmma16(a, Be1, acc[1][ct]);
    }
    __syncthreads();
  }

  // ---- epilogue: out[b][c][q] = acc[c,q] / l[q] ----
#pragma unroll
  for (int qt = 0; qt < 2; ++qt) {
    float rl = 1.0f / l[qt];
    int q = q0 + qt * 16 + lan;
    float* ob = out + (size_t)b * NC * NP + q;
#pragma unroll
    for (int ct = 0; ct < 4; ++ct)
#pragma unroll
      for (int j = 0; j < 8; ++j) {
        int c = ct * 16 + j + 8 * khi;   // D layout: M = j + 8*(lane>=16)
        ob[c * NP] = acc[qt][ct][j] * rl;
      }
  }
}

extern "C" void kernel_launch(void* const* d_in, const int* in_sizes, int n_in,
                              void* d_out, int out_size, void* d_ws, size_t ws_size,
                              hipStream_t stream) {
  const float* F = (const float*)d_in[0];
  // d_in[1] (masks) only affects module state in the reference, not this output.
  float* out = (float*)d_out;

  const size_t halfmat = (size_t)NB * NP * NC * sizeof(_Float16);     // 2 MB each
  _Float16* kernH  = (_Float16*)d_ws;
  _Float16* kernT  = (_Float16*)((char*)d_ws + halfmat);
  _Float16* fpoolT = (_Float16*)((char*)d_ws + 2 * halfmat);

  prep_kern<<<(NB * NP) / 256, 256, 0, stream>>>(F, kernH, kernT);
  prep_pool<<<(NB * NC * NP) / 256, 256, 0, stream>>>(F, fpoolT);
  attn_main<<<NB * 32, 128, 0, stream>>>(kernH, kernT, fpoolT, out);
}
